// GPT2_63264868270320
// MI455X (gfx1250) — compile-verified, matrix operand order checked by reading, not run on hardware
//
#include <hip/hip_runtime.h>

// ---------------------------------------------------------------- constants
#define N_LAYERS 12
#define N_HEADS  12
#define D_MODEL  768
#define VOCAB    50257
#define BB       4
#define TT       1024
#define BT       (BB * TT)          // 4096 rows
#define HDIM     (D_MODEL / N_HEADS) // 64

typedef __attribute__((ext_vector_type(16))) __bf16 v16bf;
typedef __attribute__((ext_vector_type(8)))  __bf16 v8bf;
typedef __attribute__((ext_vector_type(8)))  float  v8f;
typedef unsigned int u32;
typedef __attribute__((ext_vector_type(4)))  u32    u32x4;
typedef __attribute__((ext_vector_type(8)))  u32    u32x8;

// ---------------------------------------------------------------- helpers
__device__ __forceinline__ unsigned short f2bf(float f) {
    unsigned int u = __float_as_uint(f);
    unsigned int r = u + 0x7FFFu + ((u >> 16) & 1u);   // round-to-nearest-even
    return (unsigned short)(r >> 16);
}

__device__ __forceinline__ float gelu_tanh(float x) {
    float x3 = x * x * x;
    return 0.5f * x * (1.f + tanhf(0.7978845608028654f * (x + 0.044715f * x3)));
}

__device__ __forceinline__ float waveSum(float v) {
#pragma unroll
    for (int off = 16; off > 0; off >>= 1) v += __shfl_xor(v, off, 32);
    return v;
}
__device__ __forceinline__ float waveMax(float v) {
#pragma unroll
    for (int off = 16; off > 0; off >>= 1) v = fmaxf(v, __shfl_xor(v, off, 32));
    return v;
}

// CDNA5 async global->LDS copy (16B per lane), tracked by ASYNCcnt.
__device__ __forceinline__ void asyncCopy16(void* ldsPtr, const void* gsrc) {
    unsigned ldsOff = (unsigned)(size_t)ldsPtr;
    asm volatile("global_load_async_to_lds_b128 %0, %1, off"
                 :: "v"(ldsOff), "v"(gsrc)
                 : "memory");
}
__device__ __forceinline__ void waitAsync() {
    asm volatile("s_wait_asynccnt 0" ::: "memory");
}

// ---------------------------------------------------------------- TDM 2D tile load
// Tensor Data Mover: DMA a [tileRows x rowElems] bf16 tile (row stride
// strideElems) from global into LDS at ldsAddr, padding +16B after every 64B
// row (matches LSTR=40) and zero-filling rows >= tensorRows (N-tail).
// D# layout per CDNA5 ISA (group0 128b, group1 256b).
__device__ __forceinline__ void tdmLoad2D(unsigned ldsAddr, const void* gsrc,
                                          u32 tileRows, u32 tensorRows,
                                          u32 rowElems, u32 strideElems) {
    unsigned long long ga = (unsigned long long)(size_t)gsrc;
    u32x4 g0;
    g0[0] = 1u;                                            // count=1, user mode
    g0[1] = ldsAddr;                                       // lds_addr (bytes)
    g0[2] = (u32)ga;                                       // global_addr[31:0]
    g0[3] = (u32)((ga >> 32) & 0x01FFFFFFu) | (2u << 30);  // addr[56:32] | type=2
    u32x8 g1;
    // data_size=1 (2B), pad_enable=1, pad_interval=3 (16 DW = 64B), pad_amount=3 (4 DW = 16B)
    g1[0] = (1u << 16) | (1u << 20) | (3u << 22) | (3u << 25);
    g1[1] = (rowElems & 0xFFFFu) << 16;                    // tensor_dim0[15:0] @ [63:48]
    g1[2] = ((rowElems >> 16) & 0xFFFFu) | ((tensorRows & 0xFFFFu) << 16); // dim0 hi | dim1 lo
    g1[3] = ((tensorRows >> 16) & 0xFFFFu) | ((rowElems & 0xFFFFu) << 16); // dim1 hi | tile_dim0
    g1[4] = tileRows & 0xFFFFu;                            // tile_dim1 | tile_dim2=0
    g1[5] = strideElems;                                   // tensor_dim0_stride[31:0]
    g1[6] = 0u;                                            // stride hi | dim1_stride lo
    g1[7] = 0u;
    asm volatile("tensor_load_to_lds %0, %1" :: "s"(g0), "s"(g1) : "memory");
}

// ---------------------------------------------------------------- cvt f32 -> bf16
__global__ __launch_bounds__(256) void k_cvt_bf16(const float* __restrict__ in,
                                                  unsigned short* __restrict__ out,
                                                  size_t n) {
    size_t i = (size_t)blockIdx.x * blockDim.x + threadIdx.x;
    size_t stride = (size_t)gridDim.x * blockDim.x;
    for (; i < n; i += stride) out[i] = f2bf(in[i]);
}

// batched transpose + cvt: in [L][R, C] f32 -> out [L][C, R] bf16 (blockIdx.z = layer)
// R, C are multiples of 32 for every weight we transpose. Tile staged into LDS
// with async global->LDS copies (16B per lane).
__global__ __launch_bounds__(256) void k_transpose_cvt(const float* __restrict__ in,
                                                       unsigned short* __restrict__ out,
                                                       int R, int C) {
    __shared__ float tile[32][36];   // +4 floats pad: 144B rows, 16B aligned
    size_t plane = (size_t)R * C;
    const float* inL = in + plane * blockIdx.z;
    unsigned short* outL = out + plane * blockIdx.z;
    int r0 = blockIdx.y * 32, c0 = blockIdx.x * 32;
    int tx = threadIdx.x & 31, ty = threadIdx.x >> 5;

    int row = threadIdx.x >> 3;          // 0..31
    int c4  = (threadIdx.x & 7) * 4;     // 0,4,...,28
    asyncCopy16(&tile[row][c4], inL + (size_t)(r0 + row) * C + c0 + c4);
    waitAsync();
    __syncthreads();
#pragma unroll
    for (int i = ty; i < 32; i += 8) {
        int c = c0 + i, r = r0 + tx;     // out[c][r]
        outL[(size_t)c * R + r] = f2bf(tile[tx][i]);
    }
}

// ---------------------------------------------------------------- embedding
__global__ __launch_bounds__(256) void k_embed(const int* __restrict__ inp,
                                               const float* __restrict__ wte,
                                               const float* __restrict__ wpe,
                                               float* __restrict__ X) {
    int bt = blockIdx.x;
    int t = bt % TT;
    int tok = inp[bt];
    for (int c = threadIdx.x; c < D_MODEL; c += 256)
        X[(size_t)bt * D_MODEL + c] = wte[(size_t)tok * D_MODEL + c] + wpe[(size_t)t * D_MODEL + c];
}

// ---------------------------------------------------------------- layernorm -> bf16
__global__ __launch_bounds__(256) void k_layernorm(const float* __restrict__ x,
                                                   const float* __restrict__ w,
                                                   const float* __restrict__ b,
                                                   unsigned short* __restrict__ out) {
    __shared__ float ws1[8], ws2[8], bc[2];
    int row = blockIdx.x;
    const float* xr = x + (size_t)row * D_MODEL;
    float s = 0.f, s2 = 0.f;
    for (int c = threadIdx.x; c < D_MODEL; c += 256) {
        float v = xr[c];
        s += v; s2 += v * v;
    }
    s = waveSum(s); s2 = waveSum(s2);
    int wv = threadIdx.x >> 5, ln = threadIdx.x & 31;
    if (ln == 0) { ws1[wv] = s; ws2[wv] = s2; }
    __syncthreads();
    if (threadIdx.x < 32) {
        s  = (ln < 8) ? ws1[ln] : 0.f;
        s2 = (ln < 8) ? ws2[ln] : 0.f;
#pragma unroll
        for (int off = 4; off > 0; off >>= 1) { s += __shfl_xor(s, off, 32); s2 += __shfl_xor(s2, off, 32); }
        if (ln == 0) {
            float mean = s / D_MODEL;
            float var  = s2 / D_MODEL - mean * mean;
            bc[0] = mean;
            bc[1] = rsqrtf(var + 1e-5f);
        }
    }
    __syncthreads();
    float mean = bc[0], rstd = bc[1];
    for (int c = threadIdx.x; c < D_MODEL; c += 256) {
        float v = (xr[c] - mean) * rstd * w[c] + b[c];
        out[(size_t)row * D_MODEL + c] = f2bf(v);
    }
}

// ---------------------------------------------------------------- WMMA GEMM
// C[M,N] = act(A[M,K] * Bw^T + bias) + resid
// A  [M,K] bf16 row-major; Bw [N,K] bf16 row-major (pre-transposed weights).
// 256 thr = 8 waves; 128x128 tile, K-step 32; double-buffered LDS filled by
// the Tensor Data Mover (TENSORcnt), fragments via ds_load_b128 pairs.
#define BM 128
#define BN 128
#define BK 32
#define LSTR 40   // bf16 units per LDS row (32 + 8 pad -> 80B; TDM pad reproduces this)

__global__ __launch_bounds__(256) void k_gemm_bf16(
    const unsigned short* __restrict__ A,
    const unsigned short* __restrict__ Bw,
    const float* __restrict__ bias,      // [N] or nullptr
    const float* __restrict__ resid,     // [M,N] or nullptr
    float* __restrict__ outF,            // [M,N] or nullptr
    unsigned short* __restrict__ outB,   // [M,N] or nullptr
    int N, int K, int doGelu)
{
    __shared__ unsigned short As[2][BM * LSTR];
    __shared__ unsigned short Bs[2][BN * LSTR];

    const int tid  = threadIdx.x;
    const int lane = tid & 31;
    const int wave = tid >> 5;
    const int wm   = wave >> 2;          // 0..1  -> 64 rows
    const int wn   = wave & 3;           // 0..3  -> 32 cols
    const int tileM = blockIdx.y * BM;
    const int tileN = blockIdx.x * BN;
    const int half = lane >> 4;          // K-group select
    const int l16  = lane & 15;
    const int koff = half * 8;           // bf16 units

    const int kTiles = K / BK;
    const u32 bRows = (u32)(N - tileN) < (u32)BN ? (u32)(N - tileN) : (u32)BN;

    // TDM staging: wave 0 issues two descriptor loads (A tile, B tile);
    // DMA zero-fills the lm_head N-tail rows via tensor_dim1.
    auto stage = [&](int kt, int buf) {
        if (wave == 0) {
            const int k0 = kt * BK;
            tdmLoad2D((unsigned)(size_t)&As[buf][0],
                      A + (size_t)tileM * K + k0,
                      (u32)BM, (u32)BM, (u32)BK, (u32)K);
            tdmLoad2D((unsigned)(size_t)&Bs[buf][0],
                      Bw + (size_t)tileN * K + k0,
                      (u32)BN, bRows, (u32)BK, (u32)K);
        }
    };

    v8f acc[4][2];
#pragma unroll
    for (int i = 0; i < 4; ++i)
#pragma unroll
        for (int j = 0; j < 2; ++j)
#pragma unroll
            for (int e = 0; e < 8; ++e) acc[i][j][e] = 0.f;

    // prologue: stage tile 0
    stage(0, 0);
    if (wave == 0) __builtin_amdgcn_s_wait_tensorcnt(0);
    __syncthreads();

    for (int kt = 0; kt < kTiles; ++kt) {
        const int cur = kt & 1;
        if (kt + 1 < kTiles) stage(kt + 1, cur ^ 1);   // overlap next tile DMA

        // ---- fragments from LDS (2x ds_load_b128 each, CDNA5 16-bit A layout) ----
        v16bf afr[4], bfr[2];
#pragma unroll
        for (int mt = 0; mt < 4; ++mt) {
            int row = wm * 64 + mt * 16 + l16;
            v8bf lo = *reinterpret_cast<const v8bf*>(&As[cur][row * LSTR + koff]);
            v8bf hi = *reinterpret_cast<const v8bf*>(&As[cur][row * LSTR + koff + 16]);
            afr[mt] = __builtin_shufflevector(lo, hi, 0,1,2,3,4,5,6,7,8,9,10,11,12,13,14,15);
        }
#pragma unroll
        for (int nt = 0; nt < 2; ++nt) {
            int col = wn * 32 + nt * 16 + l16;
            v8bf lo = *reinterpret_cast<const v8bf*>(&Bs[cur][col * LSTR + koff]);
            v8bf hi = *reinterpret_cast<const v8bf*>(&Bs[cur][col * LSTR + koff + 16]);
            bfr[nt] = __builtin_shufflevector(lo, hi, 0,1,2,3,4,5,6,7,8,9,10,11,12,13,14,15);
        }
        // ---- 8 WMMAs ----
#pragma unroll
        for (int mt = 0; mt < 4; ++mt)
#pragma unroll
            for (int nt = 0; nt < 2; ++nt)
                acc[mt][nt] = __builtin_amdgcn_wmma_f32_16x16x32_bf16(
                    false, afr[mt], false, bfr[nt], (short)0, acc[mt][nt], false, false);

        if (wave == 0) __builtin_amdgcn_s_wait_tensorcnt(0);  // next buffer resident
        __syncthreads();    // everyone done reading cur + DMA data visible
    }

    // ---- epilogue: bias -> gelu -> residual -> fp32/bf16 store ----
#pragma unroll
    for (int mt = 0; mt < 4; ++mt) {
#pragma unroll
        for (int nt = 0; nt < 2; ++nt) {
            int gn = tileN + wn * 32 + nt * 16 + l16;
            if (gn >= N) continue;
#pragma unroll
            for (int r = 0; r < 8; ++r) {
                int gm = tileM + wm * 64 + mt * 16 + half * 8 + r;
                float v = acc[mt][nt][r];
                if (bias) v += bias[gn];
                if (doGelu) v = gelu_tanh(v);
                size_t idx = (size_t)gm * N + gn;
                if (resid) v += resid[idx];
                if (outF) outF[idx] = v;
                if (outB) outB[idx] = f2bf(v);
            }
        }
    }
}

// ---------------------------------------------------------------- attention (flash, 1 wave / query row)
__global__ __launch_bounds__(256) void k_attn(const float* __restrict__ qkv, // [BT, 3C]
                                              float* __restrict__ ctx)       // [BT, C]
{
    int gwave = (int)((blockIdx.x * 256u + threadIdx.x) >> 5);
    int lane = threadIdx.x & 31;
    int t  = gwave % TT;
    int bh = gwave / TT;
    int h  = bh % N_HEADS;
    int b  = bh / N_HEADS;

    const size_t rowQ = (size_t)(b * TT + t) * (3 * D_MODEL);
    float q0 = qkv[rowQ + h * HDIM + lane];
    float q1 = qkv[rowQ + h * HDIM + lane + 32];
    const float scale = 0.125f; // 1/sqrt(64)

    float m = -3.0e38f, l = 0.f, a0 = 0.f, a1 = 0.f;
    for (int j = 0; j <= t; ++j) {
        const size_t rowK = (size_t)(b * TT + j) * (3 * D_MODEL) + D_MODEL + h * HDIM;
        float s = q0 * qkv[rowK + lane] + q1 * qkv[rowK + lane + 32];
        s = waveSum(s) * scale;
        float mNew = fmaxf(m, s);
        float corr = __expf(m - mNew);
        float p    = __expf(s - mNew);
        const size_t rowV = rowK + D_MODEL;
        a0 = a0 * corr + p * qkv[rowV + lane];
        a1 = a1 * corr + p * qkv[rowV + lane + 32];
        l  = l * corr + p;
        m  = mNew;
    }
    float inv = 1.f / l;
    size_t o = (size_t)(b * TT + t) * D_MODEL + h * HDIM;
    ctx[o + lane]      = a0 * inv;
    ctx[o + lane + 32] = a1 * inv;
}

// ---------------------------------------------------------------- log-softmax NLL per row
__global__ __launch_bounds__(256) void k_row_nll(const float* __restrict__ logits,
                                                 const int* __restrict__ target,
                                                 float* __restrict__ rownll) {
    __shared__ float red[8], bc;
    int row = blockIdx.x;
    const float* lr = logits + (size_t)row * VOCAB;
    int wv = threadIdx.x >> 5, ln = threadIdx.x & 31;

    float mx = -3.0e38f;
    for (int c = threadIdx.x; c < VOCAB; c += 256) mx = fmaxf(mx, lr[c]);
    mx = waveMax(mx);
    if (ln == 0) red[wv] = mx;
    __syncthreads();
    if (threadIdx.x < 32) {
        mx = (ln < 8) ? red[ln] : -3.0e38f;
        mx = waveMax(mx);
        if (ln == 0) bc = mx;
    }
    __syncthreads();
    mx = bc;

    float se = 0.f;
    for (int c = threadIdx.x; c < VOCAB; c += 256) se += __expf(lr[c] - mx);
    se = waveSum(se);
    __syncthreads();
    if (ln == 0) red[wv] = se;
    __syncthreads();
    if (threadIdx.x == 0) {
        float s = 0.f;
#pragma unroll
        for (int i = 0; i < 8; ++i) s += red[i];
        int tgt = target[row];
        rownll[row] = -(lr[tgt] - mx - __logf(s));
    }
}

__global__ __launch_bounds__(256) void k_mean(const float* __restrict__ rownll,
                                              float* __restrict__ out, int n) {
    __shared__ float red[8];
    float s = 0.f;
    for (int i = threadIdx.x; i < n; i += 256) s += rownll[i];
    s = waveSum(s);
    int wv = threadIdx.x >> 5, ln = threadIdx.x & 31;
    if (ln == 0) red[wv] = s;
    __syncthreads();
    if (threadIdx.x == 0) {
        float t = 0.f;
#pragma unroll
        for (int i = 0; i < 8; ++i) t += red[i];
        out[0] = t / (float)n;
    }
}

// ---------------------------------------------------------------- launch
static inline size_t alignUp(size_t v, size_t a) { return (v + a - 1) & ~(a - 1); }

extern "C" void kernel_launch(void* const* d_in, const int* in_sizes, int n_in,
                              void* d_out, int out_size, void* d_ws, size_t ws_size,
                              hipStream_t stream) {
    (void)in_sizes; (void)n_in; (void)out_size; (void)ws_size;

    const int*   inp    = (const int*)  d_in[0];
    const int*   target = (const int*)  d_in[1];
    const float* wte    = (const float*)d_in[2];
    const float* wpe    = (const float*)d_in[3];
    const float* ln1_w  = (const float*)d_in[4];
    const float* ln1_b  = (const float*)d_in[5];
    const float* attn_w = (const float*)d_in[6];
    const float* attn_b = (const float*)d_in[7];
    const float* proj_w = (const float*)d_in[8];
    const float* proj_b = (const float*)d_in[9];
    const float* ln2_w  = (const float*)d_in[10];
    const float* ln2_b  = (const float*)d_in[11];
    const float* fc_w   = (const float*)d_in[12];
    const float* fc_b   = (const float*)d_in[13];
    const float* fc2_w  = (const float*)d_in[14];
    const float* fc2_b  = (const float*)d_in[15];
    const float* lnf_w  = (const float*)d_in[16];
    const float* lnf_b  = (const float*)d_in[17];

    float* logits = (float*)d_out;                              // [BT, VOCAB]
    float* loss   = logits + (size_t)BT * VOCAB;                // [1]

    // ---- workspace carve ----
    char* p = (char*)d_ws;
    size_t off = 0;
    auto carve = [&](size_t bytes) { void* r = p + off; off = alignUp(off + bytes, 256); return r; };

    // all weights stored transposed: [N, K] bf16 row-major
    unsigned short* wq_bf   = (unsigned short*)carve((size_t)N_LAYERS * D_MODEL * 3 * D_MODEL * 2);
    unsigned short* wp_bf   = (unsigned short*)carve((size_t)N_LAYERS * D_MODEL * D_MODEL * 2);
    unsigned short* wfc_bf  = (unsigned short*)carve((size_t)N_LAYERS * D_MODEL * 4 * D_MODEL * 2);
    unsigned short* wfc2_bf = (unsigned short*)carve((size_t)N_LAYERS * 4 * D_MODEL * D_MODEL * 2);
    unsigned short* wte_bf  = (unsigned short*)carve((size_t)VOCAB * D_MODEL * 2); // [VOCAB,768] = [N,K]
    float*          X       = (float*)         carve((size_t)BT * D_MODEL * 4);
    unsigned short* act_bf  = (unsigned short*)carve((size_t)BT * D_MODEL * 2);
    unsigned short* hfc_bf  = (unsigned short*)carve((size_t)BT * 4 * D_MODEL * 2);
    float*          QKV     = (float*)         carve((size_t)BT * 3 * D_MODEL * 4);
    float*          CTX     = (float*)         carve((size_t)BT * D_MODEL * 4);
    float*          rownll  = (float*)         carve((size_t)BT * 4);

    // ---- weight prep: transpose+cvt each [K,N] -> [N,K] bf16; wte needs cvt only ----
    k_transpose_cvt<<<dim3(3 * D_MODEL / 32, D_MODEL / 32, N_LAYERS), 256, 0, stream>>>(
        attn_w, wq_bf, D_MODEL, 3 * D_MODEL);
    k_transpose_cvt<<<dim3(D_MODEL / 32, D_MODEL / 32, N_LAYERS), 256, 0, stream>>>(
        proj_w, wp_bf, D_MODEL, D_MODEL);
    k_transpose_cvt<<<dim3(4 * D_MODEL / 32, D_MODEL / 32, N_LAYERS), 256, 0, stream>>>(
        fc_w, wfc_bf, D_MODEL, 4 * D_MODEL);
    k_transpose_cvt<<<dim3(D_MODEL / 32, 4 * D_MODEL / 32, N_LAYERS), 256, 0, stream>>>(
        fc2_w, wfc2_bf, 4 * D_MODEL, D_MODEL);
    k_cvt_bf16<<<4096, 256, 0, stream>>>(wte, wte_bf, (size_t)VOCAB * D_MODEL);

    // ---- embedding ----
    k_embed<<<BT, 256, 0, stream>>>(inp, wte, wpe, X);

    const dim3 blk(256);
    const int gy = BT / BM; // 32

    for (int l = 0; l < N_LAYERS; ++l) {
        // ln1 -> bf16
        k_layernorm<<<BT, blk, 0, stream>>>(X, ln1_w + l * D_MODEL, ln1_b + l * D_MODEL, act_bf);
        // qkv = ln1 @ Wqkv + b
        k_gemm_bf16<<<dim3(3 * D_MODEL / BN, gy), blk, 0, stream>>>(
            act_bf, wq_bf + (size_t)l * D_MODEL * 3 * D_MODEL,
            attn_b + (size_t)l * 3 * D_MODEL, nullptr, QKV, nullptr,
            3 * D_MODEL, D_MODEL, 0);
        // flash attention
        k_attn<<<(BB * N_HEADS * TT) / 8, blk, 0, stream>>>(QKV, CTX);
        k_cvt_bf16<<<4096, blk, 0, stream>>>(CTX, act_bf, (size_t)BT * D_MODEL);
        // x = x + ctx @ Wproj + b   (residual fused, in-place)
        k_gemm_bf16<<<dim3(D_MODEL / BN, gy), blk, 0, stream>>>(
            act_bf, wp_bf + (size_t)l * D_MODEL * D_MODEL,
            proj_b + (size_t)l * D_MODEL, X, X, nullptr,
            D_MODEL, D_MODEL, 0);
        // ln2 -> bf16
        k_layernorm<<<BT, blk, 0, stream>>>(X, ln2_w + l * D_MODEL, ln2_b + l * D_MODEL, act_bf);
        // h = gelu(ln2 @ Wfc + b) -> bf16
        k_gemm_bf16<<<dim3(4 * D_MODEL / BN, gy), blk, 0, stream>>>(
            act_bf, wfc_bf + (size_t)l * D_MODEL * 4 * D_MODEL,
            fc_b + (size_t)l * 4 * D_MODEL, nullptr, nullptr, hfc_bf,
            4 * D_MODEL, D_MODEL, 1);
        // x = x + h @ Wfc2 + b
        k_gemm_bf16<<<dim3(D_MODEL / BN, gy), blk, 0, stream>>>(
            hfc_bf, wfc2_bf + (size_t)l * 4 * D_MODEL * D_MODEL,
            fc2_b + (size_t)l * D_MODEL, X, X, nullptr,
            D_MODEL, 4 * D_MODEL, 0);
    }

    // final LN + tied lm_head (wte_bf is already [N,K]) -> logits (d_out)
    k_layernorm<<<BT, blk, 0, stream>>>(X, lnf_w, lnf_b, act_bf);
    k_gemm_bf16<<<dim3((VOCAB + BN - 1) / BN, gy), blk, 0, stream>>>(
        act_bf, wte_bf, nullptr, nullptr, logits, nullptr,
        VOCAB, D_MODEL, 0);

    // loss
    k_row_nll<<<BT, blk, 0, stream>>>(logits, target, rownll);
    k_mean<<<1, blk, 0, stream>>>(rownll, loss, BT);
}
